// SPINN_58798102282801
// MI455X (gfx1250) — compile-verified, matrix operand order checked by reading, not run on hardware
//
#include <hip/hip_runtime.h>
#include <hip/hip_bf16.h>

// ---------------------------------------------------------------------------
// SPINN TreeLSTM cell for MI455X (gfx1250, wave32, WMMA bf16)
//   B=65536 rows, H=256, I=64, 5H=1280 gate width
//   v3: 512-thread blocks (16 waves x 5 N-tiles) instead of 256x10.
//       Halves per-wave register pressure (acc 40 VGPRs) so the 5-deep
//       B-fragment ring survives register allocation, and roughly doubles
//       waves/SIMD for latency hiding on the L2 weight-fragment loads.
// ---------------------------------------------------------------------------

typedef __attribute__((ext_vector_type(16))) __bf16 v16bf;
typedef __attribute__((ext_vector_type(8)))  float  v8f;

#define BATCH   65536
#define HID     256
#define INW     64
#define NCOL    1280          // 5*H
#define NWAVE   16            // waves per block
#define NT      5             // N-tiles per wave (80 tiles / 16 waves)
#define SLD     1284          // padded LDS stride for sum_ln (bank-conflict)
#define CBD     256           // c_buf stride
#define EPSV    1e-5f

// packed-weight tile counts: (nt*KS + kt) tiles of 512 bf16 each
#define TILES_L (80*8)        // W_left  : K=256 -> 8 ksteps
#define TILES_R (80*8)        // W_right
#define TILES_I (80*2)        // W_input : K=64  -> 2 ksteps
#define PACK_OFF_L 0
#define PACK_OFF_R (TILES_L*512)
#define PACK_OFF_I ((TILES_L+TILES_R)*512)
#define PACK_TOTAL ((TILES_L+TILES_R+TILES_I)*512)   // 737280 bf16 = 1.44MB

__device__ __forceinline__ float sigm(float x) { return 1.f / (1.f + __expf(-x)); }

// ---------------------------------------------------------------------------
// Weight packing: row-major fp32 W[K][1280] -> per-(nt,kt) WMMA B fragments,
// 32 lanes x 16 bf16 contiguous per lane (mirrors the 16-bit A layout with
// N = lane%16, K pattern {h*8..h*8+7, 16+h*8..16+h*8+7}, h = lane/16).
// ---------------------------------------------------------------------------
__global__ void spinn_pack_weights(const float* __restrict__ Wl,
                                   const float* __restrict__ Wr,
                                   const float* __restrict__ Wi,
                                   __bf16* __restrict__ pack) {
  int e = blockIdx.x * 256 + threadIdx.x;
  if (e >= PACK_TOTAL) return;
  int tile = e >> 9;          // /512
  int rem  = e & 511;
  int lane = rem >> 4;
  int jj   = rem & 15;
  const float* W; int t, KS;
  if (tile < TILES_L)                 { W = Wl; t = tile;                   KS = 8; }
  else if (tile < TILES_L + TILES_R)  { W = Wr; t = tile - TILES_L;         KS = 8; }
  else                                { W = Wi; t = tile - TILES_L-TILES_R; KS = 2; }
  int nt = t / KS, kt = t % KS;
  int half = lane >> 4;
  int n    = nt * 16 + (lane & 15);
  int kl   = (jj < 8) ? (half * 8 + jj) : (16 + half * 8 + (jj - 8));
  int k    = kt * 32 + kl;
  pack[e] = (__bf16)W[(size_t)k * NCOL + n];
}

// ---------------------------------------------------------------------------
// Load a 16x32 bf16 A fragment from a row-major fp32 matrix.
// lane L (L<16): row=L,  K = {0..7, 16..23} (+kt*32); lanes 16-31: K+8.
// ---------------------------------------------------------------------------
__device__ __forceinline__ v16bf load_afrag(const float* __restrict__ A, int lda,
                                            int rowBase, int kt, int lane) {
  int m = lane & 15, half = lane >> 4;
  const float* p = A + (size_t)(rowBase + m) * lda + kt * 32 + half * 8;
  float4 f0 = *(const float4*)(p);
  float4 f1 = *(const float4*)(p + 4);
  float4 f2 = *(const float4*)(p + 16);
  float4 f3 = *(const float4*)(p + 20);
  v16bf a;
  a[0]=(__bf16)f0.x;  a[1]=(__bf16)f0.y;  a[2]=(__bf16)f0.z;  a[3]=(__bf16)f0.w;
  a[4]=(__bf16)f1.x;  a[5]=(__bf16)f1.y;  a[6]=(__bf16)f1.z;  a[7]=(__bf16)f1.w;
  a[8]=(__bf16)f2.x;  a[9]=(__bf16)f2.y;  a[10]=(__bf16)f2.z; a[11]=(__bf16)f2.w;
  a[12]=(__bf16)f3.x; a[13]=(__bf16)f3.y; a[14]=(__bf16)f3.z; a[15]=(__bf16)f3.w;
  return a;
}

__device__ __forceinline__ v16bf load_bfrag(const __bf16* __restrict__ pack,
                                            int tileIdx, int lane) {
  return *(const v16bf*)(pack + ((size_t)tileIdx * 32 + lane) * 16);
}

// ---------------------------------------------------------------------------
// One GEMM (16 x K x 1280) + bias + LayerNorm(1280), result accumulated into
// LDS sum_ln[16][SLD].  KS = K/32 ksteps.  Each of 16 waves owns NT=5 column
// tiles.  B fragments live in a 5-slot ring: slot t holds tile (t,kt) and is
// refilled with (t,kt+1) immediately after its WMMA -> 5 loads in flight.
// ---------------------------------------------------------------------------
template <int KS>
__device__ __forceinline__ void gemm_ln(const float* __restrict__ A, int lda,
                                        const __bf16* __restrict__ packW,
                                        const float* __restrict__ bias,
                                        float* sum_ln, float* row_sum, float* row_sq,
                                        float* meanA, float* rsqA,
                                        int rowBase, int lane, int wave, bool first) {
  int tid = threadIdx.x;
  if (tid < 16) { row_sum[tid] = 0.f; row_sq[tid] = 0.f; }
  __syncthreads();

  v8f acc[NT];
  #pragma unroll
  for (int t = 0; t < NT; t++) {
    #pragma unroll
    for (int r = 0; r < 8; r++) acc[t][r] = 0.f;
  }

  v16bf bbuf[NT];
  #pragma unroll
  for (int t = 0; t < NT; t++)
    bbuf[t] = load_bfrag(packW, (wave * NT + t) * KS + 0, lane);

  v16bf a = load_afrag(A, lda, rowBase, 0, lane);

  #pragma unroll
  for (int kt = 0; kt < KS; kt++) {
    v16bf anext;
    if (kt + 1 < KS) anext = load_afrag(A, lda, rowBase, kt + 1, lane);
    #pragma unroll
    for (int t = 0; t < NT; t++) {
      acc[t] = __builtin_amdgcn_wmma_f32_16x16x32_bf16(
          false, a, false, bbuf[t], (short)0, acc[t], false, false);
      if (kt + 1 < KS)
        bbuf[t] = load_bfrag(packW, (wave * NT + t) * KS + (kt + 1), lane);
    }
    if (kt + 1 < KS) a = anext;
  }

  // bias add + per-lane partial stats (each lane covers NT columns of rows
  // r+8*half, r=0..7)
  float s[8], q[8];
  #pragma unroll
  for (int r = 0; r < 8; r++) { s[r] = 0.f; q[r] = 0.f; }
  #pragma unroll
  for (int t = 0; t < NT; t++) {
    float bv = bias[(wave * NT + t) * 16 + (lane & 15)];
    #pragma unroll
    for (int r = 0; r < 8; r++) {
      float v = acc[t][r] + bv;
      acc[t][r] = v;
      s[r] += v; q[r] += v * v;
    }
  }
  // reduce across the 16 lanes of each half (rows r vs r+8)
  #pragma unroll
  for (int r = 0; r < 8; r++) {
    #pragma unroll
    for (int off = 8; off >= 1; off >>= 1) {
      s[r] += __shfl_xor(s[r], off, 32);
      q[r] += __shfl_xor(q[r], off, 32);
    }
  }
  int half = lane >> 4;
  if ((lane & 15) == 0) {
    #pragma unroll
    for (int r = 0; r < 8; r++) {
      atomicAdd(&row_sum[r + 8 * half], s[r]);
      atomicAdd(&row_sq[r + 8 * half], q[r]);
    }
  }
  __syncthreads();
  if (tid < 16) {
    float mean = row_sum[tid] * (1.f / (float)NCOL);
    float var  = row_sq[tid] * (1.f / (float)NCOL) - mean * mean;
    meanA[tid] = mean;
    rsqA[tid]  = rsqrtf(var + EPSV);
  }
  __syncthreads();
  // normalize and accumulate into sum_ln (each thread owns disjoint (m,col))
  #pragma unroll
  for (int r = 0; r < 8; r++) {
    int m = r + 8 * half;
    float mn = meanA[m], rq = rsqA[m];
    #pragma unroll
    for (int t = 0; t < NT; t++) {
      int col = (wave * NT + t) * 16 + (lane & 15);
      float v = (acc[t][r] - mn) * rq;
      float* dst = &sum_ln[m * SLD + col];
      if (first) *dst = v; else *dst += v;
    }
  }
}

// ---------------------------------------------------------------------------
// Main kernel: one block (512 threads, 16 waves) = 16 rows of the batch.
// ---------------------------------------------------------------------------
__global__ __launch_bounds__(512)
void spinn_cell_kernel(const float* __restrict__ h_left,
                       const float* __restrict__ c_left,
                       const float* __restrict__ h_right,
                       const float* __restrict__ c_right,
                       const float* __restrict__ feat,
                       const float* __restrict__ b_left,
                       const float* __restrict__ b_right,
                       const float* __restrict__ b_input,
                       const __bf16* __restrict__ pack,
                       float* __restrict__ out) {
  extern __shared__ float smem[];
  float* sum_ln  = smem;                      // 16*SLD
  float* c_buf   = sum_ln + 16 * SLD;         // 16*CBD
  float* row_sum = c_buf + 16 * CBD;          // 16
  float* row_sq  = row_sum + 16;              // 16
  float* meanA   = row_sq + 16;               // 16
  float* rsqA    = meanA + 16;                // 16
  float* psum    = rsqA + 16;                 // 16 rows * 8 group-waves
  float* psq     = psum + 128;                // 16*8

  int tid  = threadIdx.x;
  int lane = tid & 31;
  int wave = tid >> 5;
  int rowBase = blockIdx.x * 16;

  // Warm L2/WGP$ for the gate phase: this block's c_left/c_right slices are
  // contiguous 16KB regions; 512 threads x 32B stride covers them exactly.
  {
    const char* pl = (const char*)(c_left  + (size_t)rowBase * HID) + tid * 32;
    const char* pr = (const char*)(c_right + (size_t)rowBase * HID) + tid * 32;
    __builtin_prefetch(pl, 0, 3);
    __builtin_prefetch(pr, 0, 3);
  }

  gemm_ln<8>(h_left,  HID, pack + PACK_OFF_L, b_left,  sum_ln, row_sum, row_sq,
             meanA, rsqA, rowBase, lane, wave, true);
  gemm_ln<8>(h_right, HID, pack + PACK_OFF_R, b_right, sum_ln, row_sum, row_sq,
             meanA, rsqA, rowBase, lane, wave, false);
  gemm_ln<2>(feat,    INW, pack + PACK_OFF_I, b_input, sum_ln, row_sum, row_sq,
             meanA, rsqA, rowBase, lane, wave, false);
  __syncthreads();

  // -------- gate phase ----------------------------------------------------
  // group g = tid>>8 (0/1) handles rows g*8..g*8+7; j = tid&255 is the column
  int g    = tid >> 8;
  int j    = tid & 255;
  int wgrp = (tid >> 5) & 7;     // wave index within the group (0..7)
  float oval[8];
  #pragma unroll
  for (int mm = 0; mm < 8; mm++) {
    int m = g * 8 + mm;
    const float* S = sum_ln + m * SLD;
    float ag = S[j];
    float ig = S[j + 256];
    float f1 = S[j + 512];
    float f2 = S[j + 768];
    float og = S[j + 1024];
    size_t gidx = (size_t)(rowBase + m) * HID + j;
    float cl = c_left[gidx];
    float cr = c_right[gidx];
    float cv = tanhf(ag) * sigm(ig) + sigm(f1) * cl + sigm(f2) * cr;
    oval[mm] = og;
    c_buf[m * CBD + j] = cv;
    float sv = cv, qv = cv * cv;
    #pragma unroll
    for (int off = 16; off >= 1; off >>= 1) {
      sv += __shfl_xor(sv, off, 32);
      qv += __shfl_xor(qv, off, 32);
    }
    if (lane == 0) { psum[m * 8 + wgrp] = sv; psq[m * 8 + wgrp] = qv; }
  }
  __syncthreads();
  if (tid < 16) {
    float sa = 0.f, qa = 0.f;
    #pragma unroll
    for (int w = 0; w < 8; w++) { sa += psum[tid * 8 + w]; qa += psq[tid * 8 + w]; }
    float mean = sa * (1.f / (float)HID);
    float var  = qa * (1.f / (float)HID) - mean * mean;
    meanA[tid] = mean;
    rsqA[tid]  = rsqrtf(var + EPSV);
  }
  __syncthreads();

  float* out_h = out;
  float* out_c = out + (size_t)BATCH * HID;
  #pragma unroll
  for (int mm = 0; mm < 8; mm++) {
    int m = g * 8 + mm;
    size_t gidx = (size_t)(rowBase + m) * HID + j;
    float cn = (c_buf[m * CBD + j] - meanA[m]) * rsqA[m];
    out_c[gidx] = cn;
    out_h[gidx] = sigm(oval[mm]) * tanhf(cn);
  }
}

// ---------------------------------------------------------------------------
extern "C" void kernel_launch(void* const* d_in, const int* in_sizes, int n_in,
                              void* d_out, int out_size, void* d_ws, size_t ws_size,
                              hipStream_t stream) {
  const float* h_left  = (const float*)d_in[0];
  const float* c_left  = (const float*)d_in[1];
  const float* h_right = (const float*)d_in[2];
  const float* c_right = (const float*)d_in[3];
  const float* feat    = (const float*)d_in[4];
  const float* W_left  = (const float*)d_in[5];
  const float* b_left  = (const float*)d_in[6];
  const float* W_right = (const float*)d_in[7];
  const float* b_right = (const float*)d_in[8];
  const float* W_input = (const float*)d_in[9];
  const float* b_input = (const float*)d_in[10];
  float*  out  = (float*)d_out;
  __bf16* pack = (__bf16*)d_ws;   // needs PACK_TOTAL*2 = 1.44MB of workspace

  // 1) pack weights into WMMA B-fragment layout (weights stay hot in L2)
  spinn_pack_weights<<<(PACK_TOTAL + 255) / 256, 256, 0, stream>>>(
      W_left, W_right, W_input, pack);

  // 2) fused 3xGEMM + 4xLayerNorm + gates
  int smem_bytes = (16 * SLD + 16 * CBD + 4 * 16 + 2 * 128) * (int)sizeof(float);
  (void)hipFuncSetAttribute((const void*)spinn_cell_kernel,
                            hipFuncAttributeMaxDynamicSharedMemorySize, smem_bytes);
  spinn_cell_kernel<<<BATCH / 16, 512, smem_bytes, stream>>>(
      h_left, c_left, h_right, c_right, feat,
      b_left, b_right, b_input, pack, out);
}
